// AttentionLayer_26903675142357
// MI455X (gfx1250) — matrix operand order repaired and checked by reading, not compiled, not run on hardware
//
#include <hip/hip_runtime.h>

// ---------------- problem constants ----------------
#define D_MODEL 1024
#define NHEAD   16
#define DHEAD   64
#define BATCH   4
#define SEQ     1024
#define NTOK    (BATCH * SEQ)   // 4096

typedef __attribute__((ext_vector_type(16))) __bf16 v16bf;
typedef __attribute__((ext_vector_type(8)))  float  v8f;

union Frag { unsigned int u[8]; v16bf v; };
union F8   { float f[8];        v8f   v; };

#define WMMA_BF16(A, B, C) \
  __builtin_amdgcn_wmma_f32_16x16x32_bf16(false, (A), false, (B), (short)0, (C), false, false)

// f32 -> bf16 round-to-nearest-even
__device__ __forceinline__ unsigned short f2bf(float x) {
  unsigned int u = __float_as_uint(x);
  u += 0x7fffu + ((u >> 16) & 1u);
  return (unsigned short)(u >> 16);
}

// A-matrix 16x32 bf16 fragment (ISA 7.12.2): lane holds row m=lane&15;
// half = lane>>4 selects K subsets {0..7,16..23} vs {8..15,24..31}.
__device__ __forceinline__ v16bf load_a(const unsigned short* row, int half) {
  const unsigned int* p = (const unsigned int*)row;
  Frag f;
  int h4 = half * 4;
#pragma unroll
  for (int i = 0; i < 4; ++i) {
    f.u[i]     = p[h4 + i];       // K = 8*half + 2i, +1
    f.u[4 + i] = p[8 + h4 + i];   // K = 16 + 8*half + 2i, +1
  }
  return f.v;
}

// B-matrix 32x16 bf16 fragment: lane holds column n=lane&15 (col points at it);
// K order matches load_a: {8*half+0..7, 16+8*half+0..7}, row stride ld.
__device__ __forceinline__ v16bf load_b(const unsigned short* col, int ld, int half) {
  Frag f;
  int h8 = half * 8;
#pragma unroll
  for (int i = 0; i < 4; ++i) {
    f.u[i]     = (unsigned int)col[(h8 + 2 * i) * ld] | ((unsigned int)col[(h8 + 2 * i + 1) * ld] << 16);
    f.u[4 + i] = (unsigned int)col[(16 + h8 + 2 * i) * ld] | ((unsigned int)col[(16 + h8 + 2 * i + 1) * ld] << 16);
  }
  return f.v;
}

// CDNA5 async copy: 16B global -> LDS per lane, tracked by ASYNCcnt.
// VDST carries the per-lane LDS byte offset (low 32 bits of generic pointer
// per the aperture mapping), ADDR is the 64-bit global address.
__device__ __forceinline__ void async_cp16(void* lds, const void* g) {
  unsigned lds_off = (unsigned)(size_t)lds;
  asm volatile("global_load_async_to_lds_b128 %0, %1, off"
               :: "v"(lds_off), "v"(g)
               : "memory");
}
__device__ __forceinline__ void wait_async() {
  asm volatile("s_wait_asynccnt 0" ::: "memory");
}

// ---------------- kernel 1: f32 -> bf16 conversion ----------------
__global__ __launch_bounds__(256) void cvt_kernel(
    const float* __restrict__ Xf, const float* __restrict__ Xt,
    const float* __restrict__ Wq, const float* __restrict__ Wk, const float* __restrict__ Wv,
    unsigned short* __restrict__ Xfb, unsigned short* __restrict__ Xtb,
    unsigned short* __restrict__ Wqb, unsigned short* __restrict__ Wkb,
    unsigned short* __restrict__ Wvb) {
  int stride = gridDim.x * blockDim.x;
  int i0 = blockIdx.x * blockDim.x + threadIdx.x;
  for (int i = i0; i < NTOK * D_MODEL; i += stride) {
    Xfb[i] = f2bf(Xf[i]);
    Xtb[i] = f2bf(Xt[i]);
  }
  for (int i = i0; i < D_MODEL * D_MODEL; i += stride) {
    Wqb[i] = f2bf(Wq[i]);
    Wkb[i] = f2bf(Wk[i]);
    Wvb[i] = f2bf(Wv[i]);
  }
}

// ---------------- kernel 2: projection GEMM ----------------
// Y = Xb[NTOK x 1024] @ Wb[1024 x 1024] + bias, output reshaped per-head bf16.
// transpose_out == 0 : out[((b*16+h)*SEQ + s)*64 + dh]   (token-major; Q, V)
// transpose_out == 1 : out[((b*16+h)*64 + dh)*SEQ + s]   (d-major; K)
__global__ __launch_bounds__(256) void proj_kernel(
    const unsigned short* __restrict__ Xb, const unsigned short* __restrict__ Wb,
    const float* __restrict__ bias, unsigned short* __restrict__ out, int transpose_out) {
  int wave = (blockIdx.x * blockDim.x + threadIdx.x) >> 5;   // 16384 waves
  int lane = threadIdx.x & 31;
  int half = lane >> 4, ln = lane & 15;
  int tn = wave & 63;       // N tile: 64 tiles of 16
  int tm = wave >> 6;       // M tile: 256 tiles of 16

  const unsigned short* arow = Xb + (size_t)(tm * 16 + ln) * D_MODEL;
  F8 acc;
#pragma unroll
  for (int r = 0; r < 8; ++r) acc.f[r] = 0.f;

#pragma unroll 4
  for (int k = 0; k < D_MODEL; k += 32) {
    v16bf a = load_a(arow + k, half);
    v16bf b = load_b(Wb + (size_t)k * D_MODEL + tn * 16 + ln, D_MODEL, half);
    acc.v = WMMA_BF16(a, b, acc.v);
  }

  int n  = tn * 16 + ln;            // output column (fixed per lane)
  float bv = bias[n];
  int h = n >> 6, dh = n & 63;
#pragma unroll
  for (int r = 0; r < 8; ++r) {
    int m = tm * 16 + r + 8 * half; // output row
    int bb = m >> 10, s = m & 1023;
    unsigned short val = f2bf(acc.f[r] + bv);
    if (!transpose_out)
      out[((size_t)((bb * NHEAD + h) * SEQ + s)) * DHEAD + dh] = val;
    else
      out[((size_t)((bb * NHEAD + h) * DHEAD + dh)) * SEQ + s] = val;
  }
}

// ---------------- kernel 3: flash attention ----------------
// One wave handles (b,h, 16 q rows); all 8 waves of a block share (b,h).
// K/V 32-token tiles are staged once per block via CDNA5 async global->LDS
// copies, DOUBLE-BUFFERED so tile kt+1 is in flight during tile kt's WMMAs.
__global__ __launch_bounds__(256) void flash_kernel(
    const unsigned short* __restrict__ Qb, const unsigned short* __restrict__ Kt,
    const unsigned short* __restrict__ Vb, float* __restrict__ out) {
  __shared__ unsigned short Pt[8][16 * 32];    // per-wave P-tile scratch (8 KB)
  __shared__ unsigned short Ks[2][64][32];     // K tiles, d-major   (8 KB)
  __shared__ unsigned short Vs[2][32][64];     // V tiles, tok-major (8 KB)

  int tid  = threadIdx.x;
  int wv   = tid >> 5;
  int lane = tid & 31;
  int half = lane >> 4, ln = lane & 15;
  int task = blockIdx.x * 8 + wv;   // 4096 tasks
  int qt = task & 63;               // q tile within (b,h)
  int bh = task >> 6;               // identical for all 8 waves in the block

  const unsigned short* qrow = Qb + ((size_t)bh * SEQ + qt * 16 + ln) * DHEAD;
  v16bf qa0 = load_a(qrow + 0,  half);
  v16bf qa1 = load_a(qrow + 32, half);
  const unsigned short* KtB = Kt + (size_t)bh * DHEAD * SEQ;
  const unsigned short* VbB = Vb + (size_t)bh * SEQ * DHEAD;

  // per-thread staging assignment: 16B (8 bf16) chunks
  int krow = tid >> 2, kcol = (tid & 3) * 8;   // 64 rows x 4 chunks
  int vrow = tid >> 3, vcol = (tid & 7) * 8;   // 32 rows x 8 chunks

  F8 o0, o1, o2, o3;
  float m[8], l[8];
#pragma unroll
  for (int r = 0; r < 8; ++r) {
    o0.f[r] = 0.f; o1.f[r] = 0.f; o2.f[r] = 0.f; o3.f[r] = 0.f;
    m[r] = -3.0e38f; l[r] = 0.f;
  }
  const float scale = 0.125f;  // 1/sqrt(64)

  // prologue: stage tile 0 into buffer 0
  async_cp16(&Ks[0][krow][kcol], KtB + (size_t)krow * SEQ + kcol);
  async_cp16(&Vs[0][vrow][vcol], VbB + (size_t)vrow * DHEAD + vcol);

  for (int kt = 0; kt < SEQ / 32; ++kt) {
    int p = kt & 1;
    wait_async();      // tile kt's copies (issued last iteration) complete
    __syncthreads();   // visible to all waves; prior reads of buf p^1 done

    if (kt + 1 < SEQ / 32) {   // stage next tile while computing this one
      async_cp16(&Ks[p ^ 1][krow][kcol],
                 KtB + (size_t)krow * SEQ + (size_t)(kt + 1) * 32 + kcol);
      async_cp16(&Vs[p ^ 1][vrow][vcol],
                 VbB + ((size_t)(kt + 1) * 32 + vrow) * (size_t)DHEAD + vcol);
    }

    // scores: two 16x16 tiles over 32 k-tokens, contraction over d=64
    F8 sc0, sc1;
#pragma unroll
    for (int r = 0; r < 8; ++r) { sc0.f[r] = 0.f; sc1.f[r] = 0.f; }
    {
      v16bf b0 = load_b(&Ks[p][0][ln], 32, half);        // d 0..31,  tok = ln
      v16bf b1 = load_b(&Ks[p][32][ln], 32, half);       // d 32..63, tok = ln
      sc0.v = WMMA_BF16(qa0, b0, sc0.v);
      sc0.v = WMMA_BF16(qa1, b1, sc0.v);
      v16bf b2 = load_b(&Ks[p][0][16 + ln], 32, half);   // tok 16+ln
      v16bf b3 = load_b(&Ks[p][32][16 + ln], 32, half);
      sc1.v = WMMA_BF16(qa0, b2, sc1.v);
      sc1.v = WMMA_BF16(qa1, b3, sc1.v);
    }

    // online softmax (row = r + 8*half; row data lives in one lane-half)
#pragma unroll
    for (int r = 0; r < 8; ++r) {
      float s0 = sc0.f[r] * scale, s1 = sc1.f[r] * scale;
      float mx = fmaxf(s0, s1);
#pragma unroll
      for (int off = 1; off < 16; off <<= 1) mx = fmaxf(mx, __shfl_xor(mx, off, 32));
      float mn = fmaxf(m[r], mx);
      float c  = __expf(m[r] - mn);
      float p0 = __expf(s0 - mn), p1 = __expf(s1 - mn);
      float rs = p0 + p1;
#pragma unroll
      for (int off = 1; off < 16; off <<= 1) rs += __shfl_xor(rs, off, 32);
      m[r] = mn;
      l[r] = l[r] * c + rs;
      o0.f[r] *= c; o1.f[r] *= c; o2.f[r] *= c; o3.f[r] *= c;
      int row = r + 8 * half;
      Pt[wv][row * 32 + ln]      = f2bf(p0);
      Pt[wv][row * 32 + 16 + ln] = f2bf(p1);
    }

    // P (D-layout -> A-layout via LDS) @ V : contraction over 32 k-tokens
    v16bf pa = load_a(&Pt[wv][ln * 32], half);
    v16bf vb0 = load_b(&Vs[p][0][ln], 64, half);
    v16bf vb1 = load_b(&Vs[p][0][16 + ln], 64, half);
    v16bf vb2 = load_b(&Vs[p][0][32 + ln], 64, half);
    v16bf vb3 = load_b(&Vs[p][0][48 + ln], 64, half);
    o0.v = WMMA_BF16(pa, vb0, o0.v);
    o1.v = WMMA_BF16(pa, vb1, o1.v);
    o2.v = WMMA_BF16(pa, vb2, o2.v);
    o3.v = WMMA_BF16(pa, vb3, o3.v);
  }

  // epilogue: scale by 1/l, write f32 ctx [B, S, D_MODEL]
  int b = bh >> 4, h = bh & 15;
#pragma unroll
  for (int r = 0; r < 8; ++r) {
    float inv = 1.0f / l[r];
    int s = qt * 16 + r + 8 * half;
    float* op = out + ((size_t)(b * SEQ + s)) * D_MODEL + h * DHEAD;
    op[0 * 16 + ln] = o0.f[r] * inv;
    op[1 * 16 + ln] = o1.f[r] * inv;
    op[2 * 16 + ln] = o2.f[r] * inv;
    op[3 * 16 + ln] = o3.f[r] * inv;
  }
}

// ---------------- launch ----------------
extern "C" void kernel_launch(void* const* d_in, const int* in_sizes, int n_in,
                              void* d_out, int out_size, void* d_ws, size_t ws_size,
                              hipStream_t stream) {
  const float* Xf = (const float*)d_in[0];  // from_tensor
  const float* Xt = (const float*)d_in[1];  // to_tensor
  const float* Wq = (const float*)d_in[2];
  const float* bq = (const float*)d_in[3];
  const float* Wk = (const float*)d_in[4];
  const float* bk = (const float*)d_in[5];
  const float* Wv = (const float*)d_in[6];
  const float* bv = (const float*)d_in[7];
  float* out = (float*)d_out;

  unsigned short* Xfb = (unsigned short*)d_ws;                 // 4096x1024
  unsigned short* Xtb = Xfb + (size_t)NTOK * D_MODEL;          // 4096x1024
  unsigned short* Wqb = Xtb + (size_t)NTOK * D_MODEL;          // 1024x1024
  unsigned short* Wkb = Wqb + (size_t)D_MODEL * D_MODEL;
  unsigned short* Wvb = Wkb + (size_t)D_MODEL * D_MODEL;
  unsigned short* Qb  = Wvb + (size_t)D_MODEL * D_MODEL;       // [BH, SEQ, 64]
  unsigned short* KtW = Qb  + (size_t)NTOK * D_MODEL;          // [BH, 64, SEQ]
  unsigned short* VbW = KtW + (size_t)NTOK * D_MODEL;          // [BH, SEQ, 64]

  cvt_kernel<<<2048, 256, 0, stream>>>(Xf, Xt, Wq, Wk, Wv, Xfb, Xtb, Wqb, Wkb, Wvb);
  proj_kernel<<<2048, 256, 0, stream>>>(Xfb, Wqb, bq, Qb, 0);  // Q token-major
  proj_kernel<<<2048, 256, 0, stream>>>(Xtb, Wkb, bk, KtW, 1); // K d-major
  proj_kernel<<<2048, 256, 0, stream>>>(Xtb, Wvb, bv, VbW, 0); // V token-major
  flash_kernel<<<512, 256, 0, stream>>>(Qb, KtW, VbW, out);
}
